// LinearAttention1d_34900904247758
// MI455X (gfx1250) — compile-verified, matrix operand order checked by reading, not run on hardware
//
#include <hip/hip_runtime.h>
#include <hip/hip_bf16.h>
#include <math.h>

// ---------------------------------------------------------------------------
// LinearAttention1d on MI455X (gfx1250, wave32, WMMA bf16 16x16x32, f32 acc)
//   b=4, c=512, l=16384, heads=8, dh=64, hidden=512
// Memory-bound pipeline -> bf16 intermediates in HBM (halves traffic), fp32
// accumulation, double-buffered async global->LDS staging (ASYNCcnt) so the
// DMA for tile i+1 overlaps the WMMA compute on tile i.
//   0) one-time bf16 copies of x, w_qkv, w_out
//   1) qkv  = w_qkv @ x            (GEMM  M=1536 K=512  N=16384, per batch)
//   2) softmax over l on k rows    (2048 rows of 16384, fp32 math on bf16)
//   3) context = k @ v^T partials  (M=64 N=64 K=16384, 16-way K-split, async)
//   4) context reduce (deterministic) -> bf16
//   5) attended = context @ q      (GEMM  M=64  K=64   N=16384, 32 bh)
//   6) out = w_out @ attended + b  (GEMM  M=512 K=512  N=16384, fp32 out)
// ---------------------------------------------------------------------------

typedef __attribute__((ext_vector_type(16))) __bf16 v16bf;
typedef __attribute__((ext_vector_type(8)))  __bf16 v8bf;
typedef __attribute__((ext_vector_type(4)))  __bf16 v4bf;
typedef __attribute__((ext_vector_type(8)))  float  v8f;

#define LDK 40  // padded LDS row stride in bf16 elems (80B) -> conflict-free b128

static constexpr int BATCH = 4;
static constexpr int CIN   = 512;
static constexpr int L     = 16384;
static constexpr int HEADS = 8;
static constexpr int DH    = 64;
static constexpr int OQKV  = 3 * 512;   // 1536
static constexpr int KSPLIT = 16;       // context K-split (1024 per split)

__device__ __forceinline__ v4bf cvt4(float4 f) {
  v4bf r;
  r[0] = (__bf16)f.x; r[1] = (__bf16)f.y; r[2] = (__bf16)f.z; r[3] = (__bf16)f.w;
  return r;
}

// Async DMA of one 16B chunk global -> LDS (CDNA5 GLOBAL_LOAD_ASYNC_TO_LDS_B128,
// tracked by ASYNCcnt, no VGPR round-trip). vdst = LDS byte address.
__device__ __forceinline__ void async_ld_b128(const __bf16* lds_dst,
                                              const __bf16* gsrc) {
  unsigned int  l = (unsigned int)(uintptr_t)lds_dst;   // addr[31:0] = LDS offset
  unsigned long long g = (unsigned long long)(uintptr_t)gsrc;
  asm volatile("global_load_async_to_lds_b128 %0, %1, off"
               :: "v"(l), "v"(g) : "memory");
}
__device__ __forceinline__ void wait_async() {
  asm volatile("s_wait_asynccnt 0x0" ::: "memory");
}

// Load one 16x32 bf16 fragment row for this lane from an LDS tile stored as
// [row][k] with stride LDK. Layout per CDNA5 ISA (16-bit A-matrix 16x32):
//   lanes 0-15 : VGPR0-3 = K 0..7,  VGPR4-7 = K 16..23
//   lanes 16-31: VGPR0-3 = K 8..15, VGPR4-7 = K 24..31
__device__ __forceinline__ v16bf frag_ld(const __bf16* rowp, int half) {
  union { v16bf v; v8bf h[2]; } u;
  u.h[0] = *(const v8bf*)(rowp + half * 8);
  u.h[1] = *(const v8bf*)(rowp + 16 + half * 8);
  return u.v;
}

// ---------------------------------------------------------------------------
// Generic bf16 WMMA GEMM:  C[m,n] = sum_k A[m,k] * B[k,n]  (+ bias[m])
// A row-major [M,K] bf16 (K-contiguous -> async LDS staging),
// B row-major [K,N] bf16 (transposed to [n][k] during staging),
// C row-major fp32 or bf16. Block tile BM x 64; each wave owns 16 x 64.
// Double-buffered: async copy of K-step i+1 overlaps WMMA on K-step i.
// ---------------------------------------------------------------------------
template <int BM, bool BIAS, bool CBF16>
__global__ __launch_bounds__((BM / 16) * 32)
void gemm_bf16_kernel(const __bf16* __restrict__ A, const __bf16* __restrict__ Bm,
                      void* __restrict__ Cv, const float* __restrict__ bias,
                      int K, int lda, int ldb, int ldc, int zdiv,
                      long long sA1, long long sA2,
                      long long sB1, long long sB2,
                      long long sC1, long long sC2) {
  constexpr int NW = BM / 16;
  constexpr int NT = NW * 32;
  __shared__ __align__(16) __bf16 Alds[2][BM * LDK];
  __shared__ __align__(16) __bf16 Blds[2][64 * LDK];

  const int tid  = threadIdx.x;
  const int lane = tid & 31;
  const int wave = tid >> 5;
  const int half = (lane >> 4) & 1;
  const int l16  = lane & 15;

  const int n0 = blockIdx.x * 64;
  const int m0 = blockIdx.y * BM;
  const int zb = blockIdx.z / zdiv;
  const int zh = blockIdx.z % zdiv;

  const __bf16* Ab = A + (long long)zb * sA1 + (long long)zh * sA2 +
                     (long long)m0 * lda;
  const __bf16* Bb = Bm + (long long)zb * sB1 + (long long)zh * sB2 + n0;
  const long long cbase = (long long)zb * sC1 + (long long)zh * sC2 +
                          (long long)m0 * ldc + n0;

  v8f acc[4] = {v8f{}, v8f{}, v8f{}, v8f{}};

  auto stage = [&](int k0, int buf) {
    // A tile: BM x 32 bf16, K-contiguous -> pure async DMA to LDS
    #pragma unroll
    for (int idx = tid; idx < BM * 4; idx += NT) {
      const int row = idx >> 2, c8 = idx & 3;
      async_ld_b128(&Alds[buf][row * LDK + c8 * 8],
                    Ab + (long long)row * lda + k0 + c8 * 8);
    }
    // B tile: 32 x 64 bf16, transposed to [n][k]
    #pragma unroll
    for (int idx = tid; idx < 256; idx += NT) {
      const int kr = idx >> 3, nc8 = idx & 7;
      v8bf v = *(const v8bf*)(Bb + (long long)(k0 + kr) * ldb + nc8 * 8);
      const int nb = nc8 * 8;
      #pragma unroll
      for (int e = 0; e < 8; ++e) Blds[buf][(nb + e) * LDK + kr] = v[e];
    }
  };

  stage(0, 0);
  wait_async();
  __syncthreads();

  int buf = 0;
  for (int k0 = 0; k0 < K; k0 += 32) {
    if (k0 + 32 < K) stage(k0 + 32, buf ^ 1);   // DMA next tile in background

    const v16bf a = frag_ld(&Alds[buf][(wave * 16 + l16) * LDK], half);
    #pragma unroll
    for (int t = 0; t < 4; ++t) {
      const v16bf b = frag_ld(&Blds[buf][(t * 16 + l16) * LDK], half);
      acc[t] = __builtin_amdgcn_wmma_f32_16x16x32_bf16(
          false, a, false, b, (short)0, acc[t], false, false);
    }
    wait_async();      // next tile landed (mostly overlapped by wmma above)
    __syncthreads();   // all waves: next tile written, current tile consumed
    buf ^= 1;
  }

  // ---- epilogue: C/D layout = VGPR r -> row (r + 8*half), col l16
  const int crow = wave * 16 + half * 8;
  #pragma unroll
  for (int r = 0; r < 8; ++r) {
    const long long ro = cbase + (long long)(crow + r) * ldc;
    const float bv = BIAS ? bias[m0 + crow + r] : 0.0f;
    if (CBF16) {
      __bf16* C = (__bf16*)Cv;
      #pragma unroll
      for (int t = 0; t < 4; ++t) C[ro + t * 16 + l16] = (__bf16)(acc[t][r] + bv);
    } else {
      float* C = (float*)Cv;
      #pragma unroll
      for (int t = 0; t < 4; ++t) C[ro + t * 16 + l16] = acc[t][r] + bv;
    }
  }
}

// ---------------------------------------------------------------------------
// Softmax over l on bf16 k rows (in place), fp32 math. One block per row.
// ---------------------------------------------------------------------------
__global__ __launch_bounds__(256)
void softmax_rows_kernel(__bf16* __restrict__ qkv) {
  const int r = blockIdx.x;          // 0 .. BATCH*512-1
  const int b = r >> 9, d = r & 511;
  __bf16* row = qkv + ((long long)b * OQKV + 512 + d) * L;
  __shared__ float red[256];
  const int tid = threadIdx.x;

  float m = -INFINITY;
  for (int i = tid * 8; i < L; i += 256 * 8) {
    v8bf v = *(const v8bf*)(row + i);
    #pragma unroll
    for (int e = 0; e < 8; ++e) m = fmaxf(m, (float)v[e]);
  }
  red[tid] = m; __syncthreads();
  for (int off = 128; off > 0; off >>= 1) {
    if (tid < off) red[tid] = fmaxf(red[tid], red[tid + off]);
    __syncthreads();
  }
  m = red[0]; __syncthreads();

  float s = 0.0f;
  for (int i = tid * 8; i < L; i += 256 * 8) {
    v8bf v = *(const v8bf*)(row + i);
    v8bf o;
    #pragma unroll
    for (int e = 0; e < 8; ++e) {
      const float ex = __expf((float)v[e] - m);
      s += ex;
      o[e] = (__bf16)ex;
    }
    *(v8bf*)(row + i) = o;
  }
  red[tid] = s; __syncthreads();
  for (int off = 128; off > 0; off >>= 1) {
    if (tid < off) red[tid] += red[tid + off];
    __syncthreads();
  }
  const float inv = 1.0f / red[0];
  for (int i = tid * 8; i < L; i += 256 * 8) {
    v8bf v = *(const v8bf*)(row + i);
    #pragma unroll
    for (int e = 0; e < 8; ++e) v[e] = (__bf16)((float)v[e] * inv);
    *(v8bf*)(row + i) = v;
  }
}

// ---------------------------------------------------------------------------
// context partials: per (bh, split) block computes 64x64 over K-chunk of 1024.
// A = softmaxed k rows; B[k=l][n=e] = v[e][l]: both tiles are K-contiguous
// row copies -> fully async global->LDS staging, double-buffered.
// ---------------------------------------------------------------------------
__global__ __launch_bounds__(128)
void context_partial_kernel(const __bf16* __restrict__ qkv,
                            float* __restrict__ partial) {
  const int s  = blockIdx.x;         // K split 0..15
  const int bh = blockIdx.y;         // 0..31
  const int b = bh >> 3, h = bh & 7;
  const __bf16* Km = qkv + ((long long)b * OQKV + 512  + h * DH) * L;
  const __bf16* Vm = qkv + ((long long)b * OQKV + 1024 + h * DH) * L;

  __shared__ __align__(16) __bf16 Alds[2][64 * LDK];
  __shared__ __align__(16) __bf16 Blds[2][64 * LDK];

  const int tid  = threadIdx.x;
  const int lane = tid & 31;
  const int wave = tid >> 5;
  const int half = (lane >> 4) & 1;
  const int l16  = lane & 15;
  const int kbase = s * (L / KSPLIT);

  v8f acc[4] = {v8f{}, v8f{}, v8f{}, v8f{}};

  auto stage = [&](int k0, int buf) {
    #pragma unroll
    for (int idx = tid; idx < 512; idx += 128) {
      const int row = (idx >> 2) & 63, c8 = idx & 3;
      if (idx < 256)
        async_ld_b128(&Alds[buf][row * LDK + c8 * 8],
                      Km + (long long)row * L + kbase + k0 + c8 * 8);
      else
        async_ld_b128(&Blds[buf][row * LDK + c8 * 8],
                      Vm + (long long)row * L + kbase + k0 + c8 * 8);
    }
  };

  stage(0, 0);
  wait_async();
  __syncthreads();

  int buf = 0;
  for (int k0 = 0; k0 < L / KSPLIT; k0 += 32) {
    if (k0 + 32 < L / KSPLIT) stage(k0 + 32, buf ^ 1);

    const v16bf a = frag_ld(&Alds[buf][(wave * 16 + l16) * LDK], half);
    #pragma unroll
    for (int t = 0; t < 4; ++t) {
      const v16bf bf = frag_ld(&Blds[buf][(t * 16 + l16) * LDK], half);
      acc[t] = __builtin_amdgcn_wmma_f32_16x16x32_bf16(
          false, a, false, bf, (short)0, acc[t], false, false);
    }
    wait_async();
    __syncthreads();
    buf ^= 1;
  }

  float* P = partial + ((long long)bh * KSPLIT + s) * (DH * DH);
  const int crow = wave * 16 + half * 8;
  #pragma unroll
  for (int r = 0; r < 8; ++r)
    #pragma unroll
    for (int t = 0; t < 4; ++t)
      P[(crow + r) * DH + t * 16 + l16] = acc[t][r];
}

__global__ __launch_bounds__(256)
void context_reduce_kernel(const float* __restrict__ partial,
                           __bf16* __restrict__ context) {
  const int bh = blockIdx.x;
  for (int i = threadIdx.x; i < DH * DH; i += 256) {
    float s = 0.0f;
    #pragma unroll
    for (int p = 0; p < KSPLIT; ++p)
      s += partial[((long long)bh * KSPLIT + p) * (DH * DH) + i];
    context[(long long)bh * (DH * DH) + i] = (__bf16)s;
  }
}

// one-time fp32 -> bf16 conversion (vectorized x4); n must be multiple of 4
__global__ __launch_bounds__(256)
void f32_to_bf16_kernel(const float* __restrict__ src, __bf16* __restrict__ dst,
                        long long n4) {
  const long long stride = (long long)gridDim.x * 256;
  for (long long i = (long long)blockIdx.x * 256 + threadIdx.x; i < n4; i += stride) {
    float4 f = *(const float4*)(src + i * 4);
    *(v4bf*)(dst + i * 4) = cvt4(f);
  }
}

// ---------------------------------------------------------------------------
extern "C" void kernel_launch(void* const* d_in, const int* in_sizes, int n_in,
                              void* d_out, int out_size, void* d_ws, size_t ws_size,
                              hipStream_t stream) {
  const float* x     = (const float*)d_in[0];  // [4, 512, 16384]
  const float* w_qkv = (const float*)d_in[1];  // [1536, 512]
  const float* w_out = (const float*)d_in[2];  // [512, 512]
  const float* b_out = (const float*)d_in[3];  // [512]
  float* out = (float*)d_out;                  // [4, 512, 16384]

  // workspace layout
  __bf16* qkv      = (__bf16*)d_ws;                                  // 4*1536*L
  __bf16* attended = qkv + (size_t)BATCH * OQKV * L;                 // 4*512*L
  __bf16* xbf      = attended + (size_t)BATCH * 512 * L;             // 4*512*L
  __bf16* wqkvbf   = xbf + (size_t)BATCH * CIN * L;                  // 1536*512
  __bf16* woutbf   = wqkvbf + (size_t)OQKV * CIN;                    // 512*512
  __bf16* context  = woutbf + (size_t)512 * 512;                     // 32*64*64
  float*  partial  = (float*)(context + (size_t)BATCH * HEADS * DH * DH);

  // 0) one-time bf16 copies
  {
    const long long nx = (long long)BATCH * CIN * L / 4;
    f32_to_bf16_kernel<<<(unsigned)((nx + 255) / 256), 256, 0, stream>>>(x, xbf, nx);
    const long long nq = (long long)OQKV * CIN / 4;
    f32_to_bf16_kernel<<<(unsigned)((nq + 255) / 256), 256, 0, stream>>>(w_qkv, wqkvbf, nq);
    const long long no = (long long)512 * 512 / 4;
    f32_to_bf16_kernel<<<(unsigned)((no + 255) / 256), 256, 0, stream>>>(w_out, woutbf, no);
  }

  // 1) qkv = w_qkv @ x   (per batch), bf16 out
  gemm_bf16_kernel<128, false, true><<<dim3(L / 64, OQKV / 128, BATCH), 256, 0, stream>>>(
      wqkvbf, xbf, qkv, nullptr,
      /*K*/ CIN, /*lda*/ CIN, /*ldb*/ L, /*ldc*/ L, /*zdiv*/ 1,
      0LL, 0LL,
      (long long)CIN * L, 0LL,
      (long long)OQKV * L, 0LL);

  // 2) softmax over l on k rows
  softmax_rows_kernel<<<BATCH * 512, 256, 0, stream>>>(qkv);

  // 3) context partials + 4) deterministic reduce -> bf16
  context_partial_kernel<<<dim3(KSPLIT, BATCH * HEADS), 128, 0, stream>>>(qkv, partial);
  context_reduce_kernel<<<BATCH * HEADS, 256, 0, stream>>>(partial, context);

  // 5) attended = context @ q   (batched over 32 (b,h)), bf16 out
  gemm_bf16_kernel<64, false, true><<<dim3(L / 64, 1, BATCH * HEADS), 128, 0, stream>>>(
      context, qkv, attended, nullptr,
      /*K*/ DH, /*lda*/ DH, /*ldb*/ L, /*ldc*/ L, /*zdiv*/ HEADS,
      (long long)HEADS * DH * DH, (long long)DH * DH,
      (long long)OQKV * L,        (long long)DH * L,
      (long long)512 * L,         (long long)DH * L);

  // 6) out = w_out @ attended + b_out   (per batch), fp32 out
  gemm_bf16_kernel<128, true, false><<<dim3(L / 64, 512 / 128, BATCH), 256, 0, stream>>>(
      woutbf, attended, out, b_out,
      /*K*/ 512, /*lda*/ 512, /*ldb*/ L, /*ldc*/ L, /*zdiv*/ 1,
      0LL, 0LL,
      (long long)512 * L, 0LL,
      (long long)512 * L, 0LL);
}
